// Copy61CQCNN_61375082660349
// MI455X (gfx1250) — matrix operand order repaired and checked by reading, not compiled, not run on hardware
//
#include <hip/hip_runtime.h>
#include <hip/hip_bf16.h>
#include <math.h>

// ---------------------------------------------------------------------------
// Fused hybrid-quantum MLP for MI455X (gfx1250, wave32, WMMA).
//
// One kernel: x[B,252] -> L1..L4 (WMMA f16) -> 4-qubit statevector sim (VALU,
// per-lane) -> L5..L8 (WMMA f16) -> out[B,36] fp32.
//
// Each 256-thread block = 8 waves; each wave owns 16 rows end-to-end.
// All weights live in LDS as f16 (W^T layout, [N][Kpad]) so every WMMA
// B-fragment is two contiguous ds_load_b128. Activations ping-pong between
// two per-wave LDS buffers. Total dynamic LDS = 233,664 B (< 320 KB WGP LDS).
//
// Round-4 change: the round-3 software pipeline (prefetch chunk-step s+1
// before the WMMAs of chunk-step s) is now pinned with
// __builtin_amdgcn_sched_barrier(0) fences. Round 3 showed the machine
// scheduler sinking the prefetch loads back to their consumers and the
// allocator re-coalescing the double buffer into one register octet,
// reintroducing `s_wait_dscnt 0` before every steady-state WMMA. The fences
// fix the order through scheduling, forcing distinct registers and partial
// dscnt waits (LDS latency hidden under the WMMA chain).
//
// Input flattening assumption (insertion order of setup_inputs dicts):
//  0:x 1:w1 2:b1 3:g1 4:beta1 5:w2 6:b2 7:g2 8:beta2 9:w3 10:b3 11:w4 12:b4
//  13:qw 14:w5 15:b5 16:g5 17:beta5 18:w6 19:b6 20:g6 21:beta6 22:w7 23:b7
//  24:w8 25:b8
// ---------------------------------------------------------------------------

typedef __attribute__((ext_vector_type(16))) _Float16 v16h;
typedef __attribute__((ext_vector_type(8)))  float    v8f;

#define PI_F 3.14159265358979323846f

#if defined(__has_builtin)
#if __has_builtin(__builtin_amdgcn_sched_barrier)
#define SCHED_FENCE() __builtin_amdgcn_sched_barrier(0)
#endif
#endif
#ifndef SCHED_FENCE
#define SCHED_FENCE()
#endif

// LDS layout (byte offsets). Weight tiles are f16 [Npad][Kpad].
#define OFF_W1   0        // 128 x 256 halves = 65536 B
#define OFF_W2   65536    //  64 x 128 halves = 16384 B
#define OFF_W3   81920    //  32 x  64 halves =  4096 B
#define OFF_W4   86016    //  16 x  32 halves =  1024 B
#define OFF_W5   87040    //  64 x  32 halves =  4096 B
#define OFF_W6   91136    // 128 x  64 halves = 16384 B
#define OFF_W7   107520   //  64 x 128 halves = 16384 B
#define OFF_W8   123904   //  48 x  64 halves =  6144 B
#define OFF_P    130048   // 1328 floats      =  5312 B
#define OFF_ACT  135360   // 8 waves x (4096 + 2048) halves = 98304 B
#define LDS_TOTAL 233664

// Param sub-offsets (in floats) inside P
#define P_B1 0
#define P_S1 128
#define P_T1 256
#define P_B2 384
#define P_S2 448
#define P_T2 512
#define P_B3 576
#define P_B4 608
#define P_QW 624
#define P_B5 640
#define P_S5 704
#define P_T5 768
#define P_B6 832
#define P_S6 960
#define P_T6 1088
#define P_B7 1216
#define P_B8 1280

__device__ __forceinline__ v16h load_frag(const _Float16* p0, const _Float16* p1) {
    union { v16h h; float4 f4[2]; } u;
    u.f4[0] = *(const float4*)p0;   // 8 halves (ds_load_b128)
    u.f4[1] = *(const float4*)p1;   // 8 halves (ds_load_b128)
    return u.h;
}

// Convert one fp32 weight matrix [N][K] to zero-padded f16 W^T-style LDS tile
// [Npad][Kpad] (row n = output neuron, cols = K). Npad/Kpad are call-site
// constants -> divisions fold to shifts.
__device__ __forceinline__ void loadW(const float* __restrict__ W, int N, int K,
                                      _Float16* dst, int Npad, int Kpad, int tid) {
    int tot = Npad * Kpad;
    for (int i = tid; i < tot; i += 256) {
        int n = i / Kpad, k = i - n * Kpad;
        float v = (n < N && k < K) ? W[n * K + k] : 0.0f;
        dst[i] = (_Float16)v;
    }
}

__device__ __forceinline__ void loadV(const float* __restrict__ s, int N,
                                      float* d, int Npad, int tid) {
    for (int i = tid; i < Npad; i += 256) d[i] = (i < N) ? s[i] : 0.0f;
}

// BN eval scale = gamma / sqrt(1 + eps)
__device__ __forceinline__ void loadS(const float* __restrict__ g, int N,
                                      float* d, int Npad, int tid) {
    const float inv = rsqrtf(1.0f + 1e-5f);
    for (int i = tid; i < Npad; i += 256) d[i] = (i < N) ? g[i] * inv : 0.0f;
}

// One WMMA layer for this wave's 16 rows, software-pipelined with sched fences.
// KL = Kpad/32 k-steps, NT = Npad/16 n-tiles.
// MODE: 0 = bias only, 1 = bias+relu, 2 = bias+relu+BN(scale,shift)
template<int KL, int NT, int MODE>
__device__ __forceinline__ void gemm_layer(const _Float16* __restrict__ actIn, int sIn,
                                           const _Float16* __restrict__ wt, int kpad,
                                           const float* __restrict__ bias,
                                           const float* __restrict__ scl,
                                           const float* __restrict__ sft,
                                           _Float16* __restrict__ actOut, int sOut,
                                           int ll, int hi) {
    constexpr int C   = (NT > 4) ? 4 : NT;  // B fragments per chunk-step
    constexpr int NCH = NT / C;             // chunks per k-step
    constexpr int S   = KL * NCH;           // total chunk-steps

    // A fragment (16-bit A 16x32 layout): row = ll; halves 0..7 hold
    //   K = kk*32 + 8*hi .. +7, halves 8..15 hold K = kk*32 + 16 + 8*hi .. +7
    // B fragment: col N = tt*16 + ll; same K mapping as the A fragment
    const _Float16* arow = actIn + ll * sIn + 8 * hi;
    const _Float16* wcol = wt + ll * kpad + 8 * hi;

    v8f zero = {};
    v8f acc[NT];
#pragma unroll
    for (int t = 0; t < NT; ++t) acc[t] = zero;

    v16h a[2];
    v16h bb[2][C];

    // Prologue: operands for chunk-step 0.
    a[0] = load_frag(arow, arow + 16);
#pragma unroll
    for (int c = 0; c < C; ++c) {
        const _Float16* bp = wcol + (c * 16) * kpad;
        bb[0][c] = load_frag(bp, bp + 16);
    }

#pragma unroll
    for (int s = 0; s < S; ++s) {
        const int kk  = s / NCH;
        const int ch  = s - kk * NCH;
        const int cur = s & 1;
        const int nxt = cur ^ 1;

        // Prefetch chunk-step s+1 BEFORE the WMMAs of chunk-step s.
        if (s + 1 < S) {
            const int s2  = s + 1;
            const int kk2 = s2 / NCH;
            const int ch2 = s2 - kk2 * NCH;
            if (ch2 == 0) {               // entering a new k-step: new A frag
                const _Float16* ap = arow + kk2 * 32;
                a[kk2 & 1] = load_frag(ap, ap + 16);
            }
#pragma unroll
            for (int c = 0; c < C; ++c) {
                const _Float16* bp = wcol + ((ch2 * C + c) * 16) * kpad + kk2 * 32;
                bb[nxt][c] = load_frag(bp, bp + 16);
            }
        }

        // Pin the pipeline: loads of s+1 may not sink below, WMMAs of s may
        // not hoist above.
        SCHED_FENCE();

#pragma unroll
        for (int c = 0; c < C; ++c) {
            acc[ch * C + c] = __builtin_amdgcn_wmma_f32_16x16x32_f16(
                false, a[kk & 1], false, bb[cur][c], (short)0, acc[ch * C + c],
                false, false);
        }

        SCHED_FENCE();
    }

    // Epilogue: C layout VGPR r -> M = 8*hi + r, N = ll (per n-tile).
#pragma unroll
    for (int t = 0; t < NT; ++t) {
        int n = t * 16 + ll;
        float bb2 = bias[n];
        float sc = 1.0f, sh = 0.0f;
        if (MODE == 2) { sc = scl[n]; sh = sft[n]; }
#pragma unroll
        for (int r = 0; r < 8; ++r) {
            float v = acc[t][r] + bb2;
            if (MODE >= 1) v = fmaxf(v, 0.0f);
            if (MODE == 2) v = v * sc + sh;
            actOut[(8 * hi + r) * sOut + n] = (_Float16)v;
        }
    }
}

__global__ void __launch_bounds__(256, 1)
fused_qcnn_kernel(const float* __restrict__ x,
                  const float* __restrict__ w1, const float* __restrict__ b1,
                  const float* __restrict__ g1, const float* __restrict__ be1,
                  const float* __restrict__ w2, const float* __restrict__ b2,
                  const float* __restrict__ g2, const float* __restrict__ be2,
                  const float* __restrict__ w3, const float* __restrict__ b3,
                  const float* __restrict__ w4, const float* __restrict__ b4,
                  const float* __restrict__ qw,
                  const float* __restrict__ w5, const float* __restrict__ b5,
                  const float* __restrict__ g5, const float* __restrict__ be5,
                  const float* __restrict__ w6, const float* __restrict__ b6,
                  const float* __restrict__ g6, const float* __restrict__ be6,
                  const float* __restrict__ w7, const float* __restrict__ b7,
                  const float* __restrict__ w8, const float* __restrict__ b8,
                  float* __restrict__ out, int B) {
    extern __shared__ char smem[];
    _Float16* W1 = (_Float16*)(smem + OFF_W1);
    _Float16* W2 = (_Float16*)(smem + OFF_W2);
    _Float16* W3 = (_Float16*)(smem + OFF_W3);
    _Float16* W4 = (_Float16*)(smem + OFF_W4);
    _Float16* W5 = (_Float16*)(smem + OFF_W5);
    _Float16* W6 = (_Float16*)(smem + OFF_W6);
    _Float16* W7 = (_Float16*)(smem + OFF_W7);
    _Float16* W8 = (_Float16*)(smem + OFF_W8);
    float*    P  = (float*)(smem + OFF_P);
    _Float16* ACT = (_Float16*)(smem + OFF_ACT);

    const int tid  = threadIdx.x;
    const int wave = tid >> 5;
    const int lane = tid & 31;
    const int ll   = lane & 15;
    const int hi   = lane >> 4;

    // ---- one-time cooperative weight/param staging (fp32 -> f16, padded) ----
    loadW(w1, 120, 252, W1, 128, 256, tid);
    loadW(w2,  60, 120, W2,  64, 128, tid);
    loadW(w3,  28,  60, W3,  32,  64, tid);
    loadW(w4,   4,  28, W4,  16,  32, tid);
    loadW(w5,  56,   4, W5,  64,  32, tid);
    loadW(w6, 112,  56, W6, 128,  64, tid);
    loadW(w7,  56, 112, W7,  64, 128, tid);
    loadW(w8,  36,  56, W8,  48,  64, tid);

    loadV(b1, 120, P + P_B1, 128, tid);
    loadS(g1, 120, P + P_S1, 128, tid);
    loadV(be1,120, P + P_T1, 128, tid);
    loadV(b2,  60, P + P_B2,  64, tid);
    loadS(g2,  60, P + P_S2,  64, tid);
    loadV(be2, 60, P + P_T2,  64, tid);
    loadV(b3,  28, P + P_B3,  32, tid);
    loadV(b4,   4, P + P_B4,  16, tid);
    loadV(qw,  16, P + P_QW,  16, tid);
    loadV(b5,  56, P + P_B5,  64, tid);
    loadS(g5,  56, P + P_S5,  64, tid);
    loadV(be5, 56, P + P_T5,  64, tid);
    loadV(b6, 112, P + P_B6, 128, tid);
    loadS(g6, 112, P + P_S6, 128, tid);
    loadV(be6,112, P + P_T6, 128, tid);
    loadV(b7,  56, P + P_B7,  64, tid);
    loadV(b8,  36, P + P_B8,  48, tid);
    __syncthreads();

    // per-wave ping-pong activation buffers
    _Float16* bufA = ACT + wave * 6144;   // 16 x 256 halves
    _Float16* bufB = bufA + 4096;         // 16 x 128 halves

    const long rowBase = (long)blockIdx.x * 128 + wave * 16;

    // ---- stage this wave's 16 rows of x into bufA (f16, K padded to 256) ----
    for (int r = 0; r < 16; ++r) {
        long gr = rowBase + r;
        if (gr < B) {
            const float* xr = x + gr * 252;
            for (int k = lane; k < 256; k += 32)
                bufA[r * 256 + k] = (k < 252) ? (_Float16)xr[k] : (_Float16)0.0f;
        } else {
            for (int k = lane; k < 256; k += 32)
                bufA[r * 256 + k] = (_Float16)0.0f;
        }
    }
    __syncthreads();

    // ---- classical front MLP ----
    gemm_layer<8, 8, 2>(bufA, 256, W1, 256, P + P_B1, P + P_S1, P + P_T1, bufB, 128, ll, hi);
    __syncthreads();
    gemm_layer<4, 4, 2>(bufB, 128, W2, 128, P + P_B2, P + P_S2, P + P_T2, bufA, 256, ll, hi);
    __syncthreads();
    gemm_layer<2, 2, 1>(bufA, 256, W3,  64, P + P_B3, nullptr, nullptr, bufB, 128, ll, hi);
    __syncthreads();
    gemm_layer<1, 1, 0>(bufB, 128, W4,  32, P + P_B4, nullptr, nullptr, bufA, 256, ll, hi);
    __syncthreads();

    // ---- 4-qubit statevector simulation: lane < 16 simulates row ll ----
    // flat index bits: idx = q0*8 + q1*4 + q2*2 + q3  (qubit i stride = 8>>i)
    if (lane < 16) {
        float re[16], im[16];
#pragma unroll
        for (int i = 0; i < 16; ++i) { re[i] = 0.0f; im[i] = 0.0f; }
        re[0] = 1.0f;

        const _Float16* qi = bufA + ll * 256;
        // encoding RY(x * pi/2): half-angle = x * pi/4
#pragma unroll
        for (int q = 0; q < 4; ++q) {
            float h = (float)qi[q] * (PI_F * 0.25f);
            float c = __cosf(h), s = __sinf(h);
            int st = 8 >> q;
#pragma unroll
            for (int i0 = 0; i0 < 16; ++i0) {
                if (i0 & st) continue;
                int i1 = i0 | st;
                float r0 = re[i0], m0 = im[i0], r1 = re[i1], m1 = im[i1];
                re[i0] = c * r0 - s * r1;  im[i0] = c * m0 - s * m1;
                re[i1] = s * r0 + c * r1;  im[i1] = s * m0 + c * m1;
            }
        }

        const float* qwp = P + P_QW;   // [2][4][2]
#pragma unroll
        for (int l = 0; l < 2; ++l) {
#pragma unroll
            for (int q = 0; q < 4; ++q) {
                int st = 8 >> q;
                // RY(w0): half-angle t = 0.5*w0
                float t = 0.5f * qwp[(l * 4 + q) * 2 + 0];
                float c = __cosf(t), s = __sinf(t);
#pragma unroll
                for (int i0 = 0; i0 < 16; ++i0) {
                    if (i0 & st) continue;
                    int i1 = i0 | st;
                    float r0 = re[i0], m0 = im[i0], r1 = re[i1], m1 = im[i1];
                    re[i0] = c * r0 - s * r1;  im[i0] = c * m0 - s * m1;
                    re[i1] = s * r0 + c * r1;  im[i1] = s * m0 + c * m1;
                }
                // RZ(w1): phase e^{-+ i*phi/2} per bit
                float phi = 0.5f * qwp[(l * 4 + q) * 2 + 1];
                float cz = __cosf(phi), sz = __sinf(phi);
#pragma unroll
                for (int i = 0; i < 16; ++i) {
                    float r = re[i], m = im[i];
                    if (i & st) { re[i] = r * cz - m * sz; im[i] = r * sz + m * cz; }
                    else        { re[i] = r * cz + m * sz; im[i] = m * cz - r * sz; }
                }
            }
            // CNOT ring (0,1)(1,2)(2,3)(3,0)
#pragma unroll
            for (int p = 0; p < 4; ++p) {
                const int cqs[4] = {8, 4, 2, 1};   // control strides (qubit 0..3)
                const int tqs[4] = {4, 2, 1, 8};   // target strides
                int cs = cqs[p], ts = tqs[p];
#pragma unroll
                for (int i = 0; i < 16; ++i) {
                    if ((i & cs) && !(i & ts)) {
                        int j = i | ts;
                        float r = re[i], m = im[i];
                        re[i] = re[j]; im[i] = im[j];
                        re[j] = r;     im[j] = m;
                    }
                }
            }
        }

        // <Z_q> = sum p(idx) * (bit==0 ? +1 : -1); write + zero-pad K to 32
#pragma unroll
        for (int q = 0; q < 4; ++q) {
            int st = 8 >> q;
            float z = 0.0f;
#pragma unroll
            for (int i = 0; i < 16; ++i) {
                float pr = re[i] * re[i] + im[i] * im[i];
                z += (i & st) ? -pr : pr;
            }
            bufB[ll * 128 + q] = (_Float16)z;
        }
#pragma unroll
        for (int c = 4; c < 32; ++c) bufB[ll * 128 + c] = (_Float16)0.0f;
    }
    __syncthreads();

    // ---- classical back MLP ----
    gemm_layer<1, 4, 2>(bufB, 128, W5,  32, P + P_B5, P + P_S5, P + P_T5, bufA, 256, ll, hi);
    __syncthreads();
    gemm_layer<2, 8, 2>(bufA, 256, W6,  64, P + P_B6, P + P_S6, P + P_T6, bufB, 128, ll, hi);
    __syncthreads();
    gemm_layer<4, 4, 1>(bufB, 128, W7, 128, P + P_B7, nullptr, nullptr, bufA, 256, ll, hi);
    __syncthreads();

    // ---- L8: 64->36 (padded 48), accumulate f32, store straight to global ----
    // Small enough to load both k-steps' operands fully up front.
    {
        v8f zero = {};
        v8f acc[3];
#pragma unroll
        for (int nt = 0; nt < 3; ++nt) acc[nt] = zero;

        const _Float16* arow = bufA + ll * 256 + 8 * hi;
        const _Float16* wcol = W8 + ll * 64 + 8 * hi;

        v16h a0 = load_frag(arow, arow + 16);
        v16h a1 = load_frag(arow + 32, arow + 48);
        v16h b0[3], b1[3];
#pragma unroll
        for (int nt = 0; nt < 3; ++nt) {
            const _Float16* bp = wcol + (nt * 16) * 64;
            b0[nt] = load_frag(bp, bp + 16);
            b1[nt] = load_frag(bp + 32, bp + 48);
        }
        SCHED_FENCE();
#pragma unroll
        for (int nt = 0; nt < 3; ++nt)
            acc[nt] = __builtin_amdgcn_wmma_f32_16x16x32_f16(
                false, a0, false, b0[nt], (short)0, acc[nt], false, false);
#pragma unroll
        for (int nt = 0; nt < 3; ++nt)
            acc[nt] = __builtin_amdgcn_wmma_f32_16x16x32_f16(
                false, a1, false, b1[nt], (short)0, acc[nt], false, false);

#pragma unroll
        for (int nt = 0; nt < 3; ++nt) {
            int n = nt * 16 + ll;
            if (n < 36) {
                float bb = P[P_B8 + n];
#pragma unroll
                for (int r = 0; r < 8; ++r) {
                    long row = rowBase + 8 * hi + r;
                    if (row < B) out[row * 36 + n] = acc[nt][r] + bb;
                }
            }
        }
    }
}

extern "C" void kernel_launch(void* const* d_in, const int* in_sizes, int n_in,
                              void* d_out, int out_size, void* d_ws, size_t ws_size,
                              hipStream_t stream) {
    const float* x   = (const float*)d_in[0];
    const float* w1  = (const float*)d_in[1];
    const float* b1  = (const float*)d_in[2];
    const float* g1  = (const float*)d_in[3];
    const float* be1 = (const float*)d_in[4];
    const float* w2  = (const float*)d_in[5];
    const float* b2  = (const float*)d_in[6];
    const float* g2  = (const float*)d_in[7];
    const float* be2 = (const float*)d_in[8];
    const float* w3  = (const float*)d_in[9];
    const float* b3  = (const float*)d_in[10];
    const float* w4  = (const float*)d_in[11];
    const float* b4  = (const float*)d_in[12];
    const float* qw  = (const float*)d_in[13];
    const float* w5  = (const float*)d_in[14];
    const float* b5  = (const float*)d_in[15];
    const float* g5  = (const float*)d_in[16];
    const float* be5 = (const float*)d_in[17];
    const float* w6  = (const float*)d_in[18];
    const float* b6  = (const float*)d_in[19];
    const float* g6  = (const float*)d_in[20];
    const float* be6 = (const float*)d_in[21];
    const float* w7  = (const float*)d_in[22];
    const float* b7  = (const float*)d_in[23];
    const float* w8  = (const float*)d_in[24];
    const float* b8  = (const float*)d_in[25];
    float* out = (float*)d_out;

    const int B = in_sizes[0] / 252;
    const int blocks = (B + 127) / 128;

    fused_qcnn_kernel<<<blocks, 256, LDS_TOTAL, stream>>>(
        x, w1, b1, g1, be1, w2, b2, g2, be2, w3, b3, w4, b4, qw,
        w5, b5, g5, be5, w6, b6, g6, be6, w7, b7, w8, b8, out, B);
}
